// PointNetFeaturePropagation_446676598906
// MI455X (gfx1250) — matrix operand order repaired and checked by reading, not compiled, not run on hardware
//
#include <hip/hip_runtime.h>

// ---------------------------------------------------------------------------
// PointNet++ feature propagation for MI455X (gfx1250, wave32).
// GEMMs run on the bf16 WMMA pipe (V_WMMA_F32_16X16X32_BF16) with two
// independent accumulators per wave (32x16 output / wave) to break the
// C-operand dependency chain and halve LDS B-fragment traffic per FLOP.
// ---------------------------------------------------------------------------

typedef __attribute__((ext_vector_type(16))) __bf16 v16bf;
typedef __attribute__((ext_vector_type(8)))  __bf16 v8bf;
typedef __attribute__((ext_vector_type(8)))  float  v8f;

#define B_    8
#define N_    8192
#define S_    2048
#define D1_   128
#define D2_   256
#define KIN_  384           // D1 + D2
#define C1_   256
#define C2_   128
#define P_    (B_ * N_)     // 65536 points total

// ------------------------------- fp32 -> bf16 ------------------------------
__global__ void k_f2bf(const float* __restrict__ in, __bf16* __restrict__ out, int n) {
    int i = blockIdx.x * blockDim.x + threadIdx.x;
    if (i < n) out[i] = (__bf16)in[i];
}

// ---------------- kNN(4) + inverse-distance interp + concat ----------------
// One wave32 per query point. Lanes scan S/32 candidates each keeping a local
// sorted top-4, then a 5-step bitonic __shfl_xor merge yields the global top-4
// in every lane. Lanes then split the 256 feature channels for the gather.
#define PICK(ed, ei, ad, ai, bd_, bi_)                     \
    { bool t_ = (ad) <= (bd_);                             \
      ed = t_ ? (ad) : (bd_); ei = t_ ? (ai) : (bi_); }
#define CE(xd, xi, yd, yi)                                 \
    { if ((yd) < (xd)) { float td_ = xd; xd = yd; yd = td_;\
                         int   ti_ = xi; xi = yi; yi = ti_; } }

__global__ void k_interp_concat(const float* __restrict__ xyz1,
                                const float* __restrict__ xyz2,
                                const float* __restrict__ points1,
                                const float* __restrict__ points2,
                                __bf16* __restrict__ X)  // [P][384]
{
    int wave = blockIdx.x * (blockDim.x >> 5) + (threadIdx.x >> 5);
    int lane = threadIdx.x & 31;
    if (wave >= P_) return;
    int b = wave >> 13;          // / N_
    int n = wave & (N_ - 1);

    const float* x1b = xyz1 + (size_t)b * 3 * N_;
    float qx = x1b[n], qy = x1b[N_ + n], qz = x1b[2 * N_ + n];

    float d0 = 1e30f, d1 = 1e30f, d2 = 1e30f, d3 = 1e30f;
    int   i0 = 0, i1 = 0, i2 = 0, i3 = 0;

    const float* x2b = xyz2 + (size_t)b * 3 * S_;
    for (int s = lane; s < S_; s += 32) {
        float dx = qx - x2b[s];
        float dy = qy - x2b[S_ + s];
        float dz = qz - x2b[2 * S_ + s];
        float dist = dx * dx + dy * dy + dz * dz;
        if (dist < d3) {
            d3 = dist; i3 = s;
            CE(d2, i2, d3, i3);
            CE(d1, i1, d2, i2);
            CE(d0, i0, d1, i1);
        }
    }

    // cross-lane bitonic merge: after 5 xor steps every lane has global top-4
    #pragma unroll
    for (int off = 16; off >= 1; off >>= 1) {
        float b0 = __shfl_xor(d0, off, 32), b1 = __shfl_xor(d1, off, 32);
        float b2 = __shfl_xor(d2, off, 32), b3 = __shfl_xor(d3, off, 32);
        int   j0 = __shfl_xor(i0, off, 32), j1 = __shfl_xor(i1, off, 32);
        int   j2 = __shfl_xor(i2, off, 32), j3 = __shfl_xor(i3, off, 32);
        // keep the 4 smallest of (mine ++ reverse(theirs)) -> bitonic low half
        float e0, e1, e2, e3; int k0, k1, k2, k3;
        PICK(e0, k0, d0, i0, b3, j3);
        PICK(e1, k1, d1, i1, b2, j2);
        PICK(e2, k2, d2, i2, b1, j1);
        PICK(e3, k3, d3, i3, b0, j0);
        // sort the bitonic quad: CE(0,2) CE(1,3) CE(0,1) CE(2,3)
        CE(e0, k0, e2, k2); CE(e1, k1, e3, k3);
        CE(e0, k0, e1, k1); CE(e2, k2, e3, k3);
        d0 = e0; d1 = e1; d2 = e2; d3 = e3;
        i0 = k0; i1 = k1; i2 = k2; i3 = k3;
    }

    float w0 = 1.f / (d0 + 1e-8f), w1 = 1.f / (d1 + 1e-8f);
    float w2 = 1.f / (d2 + 1e-8f), w3 = 1.f / (d3 + 1e-8f);
    float rs = 1.f / (w0 + w1 + w2 + w3);
    w0 *= rs; w1 *= rs; w2 *= rs; w3 *= rs;

    __bf16* xr = X + (size_t)wave * KIN_;

    // copy points1 (layout [B,N,D1] == [P][D1])
    const float* p1 = points1 + (size_t)wave * D1_;
    #pragma unroll
    for (int t = 0; t < D1_ / 32; ++t) {
        int c = lane + 32 * t;
        xr[c] = (__bf16)p1[c];
    }
    // interpolate points2 (layout [B,D2,S])
    const float* f2 = points2 + (size_t)b * D2_ * S_;
    #pragma unroll
    for (int t = 0; t < D2_ / 32; ++t) {
        int c = lane + 32 * t;
        const float* fc = f2 + (size_t)c * S_;
        float acc = w0 * fc[i0] + w1 * fc[i1] + w2 * fc[i2] + w3 * fc[i3];
        xr[D1_ + c] = (__bf16)acc;
    }
}

// ------------------------- bf16 WMMA GEMM + bias ---------------------------
// Y[M][P] = W[M][K] * X[P][K]^T + bias.
// Each wave computes a 32x16 output (two 16x16 C tiles, two independent
// accumulators sharing one B fragment). 8 waves/block; a block stages
// (8*32/M) 16-point X tiles in LDS. K multiple of 32, M multiple of 32.
__global__ void k_gemm_wmma(const __bf16* __restrict__ Wbf,  // [M][K]
                            const __bf16* __restrict__ Xbf,  // [P][K]
                            const float*  __restrict__ bias, // [M]
                            float*        __restrict__ Y,    // [M][P]
                            int M, int K, int P)
{
    __shared__ __align__(32) __bf16 ldsX[8192];   // 16KB: 32x256 or 16x384

    const int wib  = threadIdx.x >> 5;
    const int lane = threadIdx.x & 31;
    const int wpb  = blockDim.x >> 5;             // waves per block (8)
    const int m32t = M >> 5;                      // waves covering M (8 or 4)

    const int waveBase  = blockIdx.x * wpb;
    const int tileNbase = (waveBase / m32t) * 16; // first N tile of block
    const int nIdx      = wib / m32t;             // staged tile used by wave
    const int tileN     = tileNbase + nIdx * 16;
    const int tileM     = ((waveBase + wib) % m32t) * 32;
    if (tileN >= P) return;

    // cooperative stage: X rows [tileNbase, tileNbase + 16*ntiles) x K
    const int rows   = 16 * (wpb / m32t);
    const int kq     = K >> 3;                    // 16B chunks per row
    const int chunks = rows * kq;
    for (int c = threadIdx.x; c < chunks; c += blockDim.x) {
        int row = c / kq;
        int off = (c - row * kq) << 3;
        *(uint4*)(&ldsX[row * K + off]) =
            *(const uint4*)(Xbf + (size_t)(tileNbase + row) * K + off);
    }
    __syncthreads();

    const int hi  = lane >> 4;                    // half-wave select
    const int l16 = lane & 15;
    const __bf16* arow0 = Wbf + (size_t)(tileM + l16) * K;
    const __bf16* arow1 = arow0 + (size_t)16 * K;
    const __bf16* brow  = &ldsX[(nIdx * 16 + l16) * K];

    v8f acc0 = {};
    v8f acc1 = {};
    for (int k0 = 0; k0 < K; k0 += 32) {
        // B fragment (shared): lanes0-15 K k0..k0+15, lanes16-31 k0+16..k0+31
        v16bf bfrag = *(const v16bf*)(brow + k0 + hi * 16);
        // A fragments (matching B): lanes0-15 K{0..15},
        //                           lanes16-31 K{16..31}
        v8bf a0lo = *(const v8bf*)(arow0 + k0 + hi * 16);
        v8bf a0hi = *(const v8bf*)(arow0 + k0 + 8 + hi * 16);
        v8bf a1lo = *(const v8bf*)(arow1 + k0 + hi * 16);
        v8bf a1hi = *(const v8bf*)(arow1 + k0 + 8 + hi * 16);
        v16bf a0 = __builtin_shufflevector(a0lo, a0hi,
                    0, 1, 2, 3, 4, 5, 6, 7, 8, 9, 10, 11, 12, 13, 14, 15);
        v16bf a1 = __builtin_shufflevector(a1lo, a1hi,
                    0, 1, 2, 3, 4, 5, 6, 7, 8, 9, 10, 11, 12, 13, 14, 15);
        acc0 = __builtin_amdgcn_wmma_f32_16x16x32_bf16(
                   false, a0, false, bfrag, (short)0, acc0, false, false);
        acc1 = __builtin_amdgcn_wmma_f32_16x16x32_bf16(
                   false, a1, false, bfrag, (short)0, acc1, false, false);
    }

    // C layout: vgpr r -> M = tile + r + hi*8, N = tileN + (lane&15)
    const int n = tileN + l16;
    #pragma unroll
    for (int r = 0; r < 8; ++r) {
        int m0 = tileM + r + hi * 8;
        int m1 = m0 + 16;
        Y[(size_t)m0 * P + n] = acc0[r] + bias[m0];
        Y[(size_t)m1 * P + n] = acc1[r] + bias[m1];
    }
}

// -------------------- BN stats: fold gamma/beta to scale/shift -------------
__global__ void k_bn_stats(const float* __restrict__ Y,   // [M][P]
                           const float* __restrict__ g,
                           const float* __restrict__ beta,
                           float* __restrict__ scale,
                           float* __restrict__ shift, int P)
{
    __shared__ float s_sum[256];
    __shared__ float s_sq[256];
    const int m = blockIdx.x;
    const float* row = Y + (size_t)m * P;
    float s = 0.f, q = 0.f;
    for (int i = threadIdx.x; i < P; i += blockDim.x) {
        float v = row[i];
        s += v; q += v * v;
    }
    s_sum[threadIdx.x] = s; s_sq[threadIdx.x] = q;
    __syncthreads();
    for (int o = blockDim.x >> 1; o > 0; o >>= 1) {
        if ((int)threadIdx.x < o) {
            s_sum[threadIdx.x] += s_sum[threadIdx.x + o];
            s_sq[threadIdx.x]  += s_sq[threadIdx.x + o];
        }
        __syncthreads();
    }
    if (threadIdx.x == 0) {
        float mean = s_sum[0] / (float)P;
        float var  = s_sq[0] / (float)P - mean * mean;
        float sc   = g[m] * rsqrtf(var + 1e-5f);
        scale[m] = sc;
        shift[m] = beta[m] - mean * sc;
    }
}

// ------------- BN + ReLU, transpose to [P][M] bf16 for next GEMM -----------
__global__ void k_bn_relu_tr(const float* __restrict__ Y,     // [M][P]
                             const float* __restrict__ scale,
                             const float* __restrict__ shift,
                             __bf16* __restrict__ X2,          // [P][M]
                             int M, int P)
{
    size_t i = (size_t)blockIdx.x * blockDim.x + threadIdx.x;
    if (i >= (size_t)M * P) return;
    int m = (int)(i / P);
    int n = (int)(i - (size_t)m * P);
    float v = Y[i] * scale[m] + shift[m];
    v = fmaxf(v, 0.f);
    X2[(size_t)n * M + m] = (__bf16)v;
}

// ---------------- BN + ReLU, final output layout [B, C2, N] ----------------
__global__ void k_bn_relu_out(const float* __restrict__ Y,    // [C2][P]
                              const float* __restrict__ scale,
                              const float* __restrict__ shift,
                              float* __restrict__ out, int P)
{
    size_t i = (size_t)blockIdx.x * blockDim.x + threadIdx.x;
    if (i >= (size_t)C2_ * P) return;
    int m = (int)(i / P);
    int p = (int)(i - (size_t)m * P);
    int b = p >> 13;             // / N_
    int n = p & (N_ - 1);
    float v = Y[i] * scale[m] + shift[m];
    v = fmaxf(v, 0.f);
    out[((size_t)b * C2_ + m) * N_ + n] = v;
}

// ---------------------------------------------------------------------------
extern "C" void kernel_launch(void* const* d_in, const int* in_sizes, int n_in,
                              void* d_out, int out_size, void* d_ws, size_t ws_size,
                              hipStream_t stream)
{
    const float* xyz1    = (const float*)d_in[0];
    const float* xyz2    = (const float*)d_in[1];
    const float* points1 = (const float*)d_in[2];
    const float* points2 = (const float*)d_in[3];
    const float* w1      = (const float*)d_in[4];
    const float* b1      = (const float*)d_in[5];
    const float* g1      = (const float*)d_in[6];
    const float* be1     = (const float*)d_in[7];
    const float* w2      = (const float*)d_in[8];
    const float* b2      = (const float*)d_in[9];
    const float* g2      = (const float*)d_in[10];
    const float* be2     = (const float*)d_in[11];
    float* out = (float*)d_out;

    char* ws = (char*)d_ws;
    size_t o = 0;
    __bf16* w1bf = (__bf16*)(ws + o); o += (size_t)C1_ * KIN_ * 2;   // 192 KiB
    __bf16* w2bf = (__bf16*)(ws + o); o += (size_t)C2_ * C1_ * 2;    //  64 KiB
    float* scale1 = (float*)(ws + o); o += C1_ * 4;
    float* shift1 = (float*)(ws + o); o += C1_ * 4;
    float* scale2 = (float*)(ws + o); o += C2_ * 4;
    float* shift2 = (float*)(ws + o); o += C2_ * 4;
    o = (o + 511) & ~(size_t)511;
    size_t oX = o;
    __bf16* X  = (__bf16*)(ws + o); o += (size_t)P_ * KIN_ * 2;      // 48 MiB
    float*  Y1 = (float*)(ws + o);  o += (size_t)C1_ * P_ * 4;       // 64 MiB
    __bf16* X2 = (__bf16*)(ws + o); o += (size_t)P_ * C1_ * 2;       // 32 MiB
    float*  Y2 = (float*)(ws + oX);  // reuse X region (X dead after GEMM1)

    // 1. repack weights to bf16
    k_f2bf<<<(C1_ * KIN_ + 255) / 256, 256, 0, stream>>>(w1, w1bf, C1_ * KIN_);
    k_f2bf<<<(C2_ * C1_ + 255) / 256, 256, 0, stream>>>(w2, w2bf, C2_ * C1_);

    // 2. kNN(4) + interpolation + concat -> X [P][384] bf16
    k_interp_concat<<<P_ / 8, 256, 0, stream>>>(xyz1, xyz2, points1, points2, X);

    // 3. layer 1: GEMM (256x384 @ 384x65536) -> stats -> BN/ReLU/transpose
    k_gemm_wmma<<<(C1_ / 32) * (P_ / 16) / 8, 256, 0, stream>>>(
        w1bf, X, b1, Y1, C1_, KIN_, P_);
    k_bn_stats<<<C1_, 256, 0, stream>>>(Y1, g1, be1, scale1, shift1, P_);
    k_bn_relu_tr<<<(int)(((size_t)C1_ * P_) / 256), 256, 0, stream>>>(
        Y1, scale1, shift1, X2, C1_, P_);

    // 4. layer 2: GEMM (128x256 @ 256x65536) -> stats -> BN/ReLU -> out
    k_gemm_wmma<<<(C2_ / 32) * (P_ / 16) / 8, 256, 0, stream>>>(
        w2bf, X2, b2, Y2, C2_, C1_, P_);
    k_bn_stats<<<C2_, 256, 0, stream>>>(Y2, g2, be2, scale2, shift2, P_);
    k_bn_relu_out<<<(int)(((size_t)C2_ * P_) / 256), 256, 0, stream>>>(
        Y2, scale2, shift2, out, P_);
}
